// ClusterLoss_25417616458033
// MI455X (gfx1250) — matrix operand order repaired and checked by reading, not compiled, not run on hardware
//
#include <hip/hip_runtime.h>
#include <hip/hip_bf16.h>

typedef __attribute__((ext_vector_type(16))) __bf16 v16bf;
typedef __attribute__((ext_vector_type(8)))  __bf16 v8bf;
typedef __attribute__((ext_vector_type(8)))  float  v8f;

#define DIMS    256
#define NCLUST  1024
#define HALF_B  16384
#define ROWS_X  65536
#define INV_EPS 20.0f   // 1/0.05

// ---------------- init: zero R (4x1024) and loss accumulator ----------------
__global__ void init_kernel(float* __restrict__ R, float* __restrict__ accum) {
  int i = blockIdx.x * blockDim.x + threadIdx.x;
  if (i < 4 * NCLUST) R[i] = 0.f;
  if (i == 0) accum[0] = 0.f;
}

// ------------- normalize: one wave32 per row of 256; emit bf16 hi/lo --------
__global__ __launch_bounds__(32) void normalize_kernel(
    const float* __restrict__ src, __bf16* __restrict__ hi, __bf16* __restrict__ lo) {
  const int row = blockIdx.x;
  const int t = threadIdx.x;
  const float* p = src + (size_t)row * DIMS;
  float v[8];
  float s = 0.f;
#pragma unroll
  for (int j = 0; j < 8; ++j) { v[j] = p[t + 32 * j]; s += v[j] * v[j]; }
#pragma unroll
  for (int o = 16; o > 0; o >>= 1) s += __shfl_xor(s, o, 32);
  const float scale = 1.0f / fmaxf(sqrtf(s), 1e-12f);
  __bf16* ph = hi + (size_t)row * DIMS;
  __bf16* pl = lo + (size_t)row * DIMS;
#pragma unroll
  for (int j = 0; j < 8; ++j) {
    float x = v[j] * scale;
    __bf16 h = (__bf16)x;
    ph[t + 32 * j] = h;
    pl[t + 32 * j] = (__bf16)(x - (float)h);
  }
}

// ---------------- WMMA GEMM: dist = sqrt(2-2*cos), fused exp column sums ----
union V16 { v16bf v; v8bf h[2]; };

__global__ __launch_bounds__(256) void gemm_dist_kernel(
    const __bf16* __restrict__ Xhi, const __bf16* __restrict__ Xlo,
    const __bf16* __restrict__ Chi, const __bf16* __restrict__ Clo,
    float* __restrict__ Dm, float* __restrict__ R) {
  const int lane  = threadIdx.x & 31;
  const int wave  = threadIdx.x >> 5;
  const int wm    = wave & 1;                  // 2 waves along M
  const int wn    = wave >> 1;                 // 4 waves along N
  const int blockM = blockIdx.y * 128;
  const int blockN = blockIdx.x * 128;
  const int waveM  = blockM + wm * 64;         // 64-row wave tile
  const int waveN  = blockN + wn * 32;         // 32-col wave tile
  const int l16  = lane & 15;
  const int aoff = (lane >> 4) * 16;           // A: K base within lane
  const int boff = (lane >> 4) * 16;           // B: K base within lane

  v8f acc[4][2];
#pragma unroll
  for (int m = 0; m < 4; ++m)
#pragma unroll
    for (int n = 0; n < 2; ++n)
#pragma unroll
      for (int r = 0; r < 8; ++r) acc[m][n][r] = 0.f;

  for (int kk = 0; kk < DIMS; kk += 32) {
    V16 ah[4], al[4];
    v16bf bh[2], bl[2];
#pragma unroll
    for (int m = 0; m < 4; ++m) {
      const size_t rb = (size_t)(waveM + m * 16 + l16) * DIMS + kk;
      ah[m].h[0] = *(const v8bf*)(Xhi + rb + aoff);
      ah[m].h[1] = *(const v8bf*)(Xhi + rb + 8 + aoff);
      al[m].h[0] = *(const v8bf*)(Xlo + rb + aoff);
      al[m].h[1] = *(const v8bf*)(Xlo + rb + 8 + aoff);
    }
#pragma unroll
    for (int n = 0; n < 2; ++n) {
      const size_t cb = (size_t)(waveN + n * 16 + l16) * DIMS + kk + boff;
      bh[n] = *(const v16bf*)(Chi + cb);
      bl[n] = *(const v16bf*)(Clo + cb);
    }
#pragma unroll
    for (int m = 0; m < 4; ++m)
#pragma unroll
      for (int n = 0; n < 2; ++n) {
        // split-bf16: hi*hi + hi*lo + lo*hi  ~ f32-accurate dot product
        acc[m][n] = __builtin_amdgcn_wmma_f32_16x16x32_bf16(
            false, ah[m].v, false, bh[n], (short)0, acc[m][n], false, false);
        acc[m][n] = __builtin_amdgcn_wmma_f32_16x16x32_bf16(
            false, ah[m].v, false, bl[n], (short)0, acc[m][n], false, false);
        acc[m][n] = __builtin_amdgcn_wmma_f32_16x16x32_bf16(
            false, al[m].v, false, bh[n], (short)0, acc[m][n], false, false);
      }
  }

  // epilogue: dist, store D, accumulate exp(dist/eps) per column into R[part]
  const int part = blockM >> 14;               // 16384 rows per sub-matrix
  float* Rp = R + part * NCLUST;
  const int mrow = (lane >> 4) * 8;            // upper half-lanes hold M+8
#pragma unroll
  for (int n = 0; n < 2; ++n) {
    const int col = waveN + n * 16 + l16;
    float cs = 0.f;
#pragma unroll
    for (int m = 0; m < 4; ++m) {
#pragma unroll
      for (int r = 0; r < 8; ++r) {
        const int row = waveM + m * 16 + mrow + r;
        const float c = acc[m][n][r];
        const float dist = sqrtf(fmaxf(2.0f - 2.0f * c, 1e-12f));
        Dm[(size_t)row * NCLUST + col] = dist;
        cs += __expf(dist * INV_EPS);
      }
    }
    cs += __shfl_xor(cs, 16, 32);              // combine lane pairs (same col)
    if (lane < 16) atomicAdd(&Rp[col], cs);
  }
}

// ---------------- logRK[i] = log(R[i] * K) ----------------------------------
__global__ void logrk_kernel(const float* __restrict__ R, float* __restrict__ logRK) {
  int i = blockIdx.x * blockDim.x + threadIdx.x;
  if (i < 4 * NCLUST) logRK[i] = logf(R[i] * (float)NCLUST);
}

// ---------------- block reduction (256 threads, wave32) ---------------------
__device__ __forceinline__ float block_sum_256(float v) {
  __shared__ float sm[8];
#pragma unroll
  for (int o = 16; o > 0; o >>= 1) v += __shfl_xor(v, o, 32);
  __syncthreads();
  if ((threadIdx.x & 31) == 0) sm[threadIdx.x >> 5] = v;
  __syncthreads();
  float r = (threadIdx.x < 8) ? sm[threadIdx.x] : 0.f;
#pragma unroll
  for (int o = 4; o > 0; o >>= 1) r += __shfl_xor(r, o, 32);
  return r;  // valid on thread 0
}

// ---------------- fused sinkhorn-CE loss pass: one block per paired row -----
__global__ __launch_bounds__(256) void loss_kernel(
    const float* __restrict__ Dm, const float* __restrict__ R,
    const float* __restrict__ logRK, float* __restrict__ accum) {
  const int i = blockIdx.x;
  const float* dz1 = Dm + (size_t)i * NCLUST;
  const float* dz2 = Dm + (size_t)(i + HALF_B) * NCLUST;
  const float* dp1 = Dm + (size_t)(i + 2 * HALF_B) * NCLUST;
  const float* dp2 = Dm + (size_t)(i + 3 * HALF_B) * NCLUST;
  const float* Rz1 = R;
  const float* Rz2 = R + NCLUST;
  const float* lrk1 = logRK + 2 * NCLUST;
  const float* lrk2 = logRK + 3 * NCLUST;

  float s11 = 0.f, s12 = 0.f, s21 = 0.f, s22 = 0.f;
  float cz1 = 0.f, cz2 = 0.f, cp1 = 0.f, cp2 = 0.f;
#pragma unroll
  for (int j = 0; j < 4; ++j) {
    const int k = threadIdx.x + 256 * j;
    const float e1 = __expf(dz1[k] * INV_EPS) / (Rz1[k] * (float)NCLUST);
    const float e2 = __expf(dz2[k] * INV_EPS) / (Rz2[k] * (float)NCLUST);
    const float l1 = dp1[k] * INV_EPS - lrk1[k];   // log(e_p1[k])
    const float l2 = dp2[k] * INV_EPS - lrk2[k];
    cz1 += e1; cz2 += e2;
    cp1 += __expf(l1); cp2 += __expf(l2);
    s11 += e1 * l1; s12 += e1 * l2; s21 += e2 * l1; s22 += e2 * l2;
  }
  s11 = block_sum_256(s11);
  s12 = block_sum_256(s12);
  s21 = block_sum_256(s21);
  s22 = block_sum_256(s22);
  cz1 = block_sum_256(cz1);
  cz2 = block_sum_256(cz2);
  cp1 = block_sum_256(cp1);
  cp2 = block_sum_256(cp2);
  if (threadIdx.x == 0) {
    // sum_k Pz*log(Pp) = S/cz - log(cp) for each of the 4 pairs
    const float term = (s11 + s12) / cz1 + (s21 + s22) / cz2
                     - 2.0f * logf(cp1) - 2.0f * logf(cp2);
    atomicAdd(accum, term);
  }
}

// ---------------- finalize: loss = -accum / (4 * bsz) -----------------------
__global__ void finalize_kernel(const float* __restrict__ accum, float* __restrict__ out) {
  if (threadIdx.x == 0 && blockIdx.x == 0) out[0] = -accum[0] / (4.0f * (float)HALF_B);
}

extern "C" void kernel_launch(void* const* d_in, const int* in_sizes, int n_in,
                              void* d_out, int out_size, void* d_ws, size_t ws_size,
                              hipStream_t stream) {
  const float* z       = (const float*)d_in[0];   // 32768 x 256
  const float* p       = (const float*)d_in[1];   // 32768 x 256
  const float* centers = (const float*)d_in[2];   // 1024  x 256
  float* out = (float*)d_out;

  char* ws = (char*)d_ws;
  __bf16* Xhi = (__bf16*)(ws);                               // 65536*256 bf16
  __bf16* Xlo = Xhi + (size_t)ROWS_X * DIMS;
  __bf16* Chi = (__bf16*)(ws + (size_t)67108864);            // 1024*256 bf16
  __bf16* Clo = Chi + (size_t)NCLUST * DIMS;
  float*  Dm  = (float*)(ws + (size_t)68157440);             // 65536*1024 f32
  float*  R   = (float*)(ws + (size_t)68157440 + 268435456); // 4*1024 f32
  float*  lRK = R + 4 * NCLUST;
  float*  acc = lRK + 4 * NCLUST;

  init_kernel<<<16, 256, 0, stream>>>(R, acc);

  normalize_kernel<<<2 * HALF_B, 32, 0, stream>>>(z, Xhi, Xlo);
  normalize_kernel<<<2 * HALF_B, 32, 0, stream>>>(
      p, Xhi + (size_t)2 * HALF_B * DIMS, Xlo + (size_t)2 * HALF_B * DIMS);
  normalize_kernel<<<NCLUST, 32, 0, stream>>>(centers, Chi, Clo);

  // 128x128 block tiles: grid (N=8, M=512); N innermost for L2 reuse of A.
  gemm_dist_kernel<<<dim3(NCLUST / 128, ROWS_X / 128), 256, 0, stream>>>(
      Xhi, Xlo, Chi, Clo, Dm, R);

  logrk_kernel<<<16, 256, 0, stream>>>(R, lRK);

  loss_kernel<<<HALF_B, 256, 0, stream>>>(Dm, R, lRK, acc);

  finalize_kernel<<<1, 1, 0, stream>>>(acc, out);
}
